// MultiHeadCrossAttention_43456479100938
// MI455X (gfx1250) — matrix operand order repaired and checked by reading, not compiled, not run on hardware
//
#include <hip/hip_runtime.h>
#include <hip/hip_bf16.h>

// ---------------------------------------------------------------------------
// MultiHeadCrossAttention for MI455X (gfx1250, wave32, WMMA f16 16x16x32)
// Pipeline:
//   1) f32 -> f16 conversion of x, y_enc, Wo ; transposed f16 of Wq/Wk/Wv
//   2) Q = X @ Wq, K = Y @ Wk (row-major f16), V^T = (Y @ Wv)^T (f16)
//      -- 32x32 register-tiled: 4 WMMAs per 8 b128 loads (2 loads/WMMA)
//   3) flash-attention per 16-row Q tile (online softmax, WMMA QK^T and PV)
//   4) out = O @ Wo^T + bo  (f32 output), 32x32 register-tiled
// ---------------------------------------------------------------------------

typedef _Float16 h8  __attribute__((ext_vector_type(8)));
typedef _Float16 h16 __attribute__((ext_vector_type(16)));
typedef float    f8  __attribute__((ext_vector_type(8)));

constexpr int Bn = 4;
constexpr int Hn = 16;
constexpr int Tn = 1024;
constexpr int Sn = 1024;
constexpr int Cn = 1024;   // n_embd
constexpr int Dn = 64;     // head size
constexpr float kScale = 0.03125f;  // C^-0.5 = 1/32

__device__ __forceinline__ h16 cat8(h8 lo, h8 hi) {
  return __builtin_shufflevector(lo, hi, 0,1,2,3,4,5,6,7,8,9,10,11,12,13,14,15);
}

__device__ __forceinline__ f8 wmma16(h16 a, h16 b, f8 c) {
  // v_wmma_f32_16x16x32_f16 : D = A(16x32) * B(32x16) + C(16x16 f32)
  return __builtin_amdgcn_wmma_f32_16x16x32_f16(false, a, false, b,
                                                (short)0, c, false, false);
}

// A-matrix (16x32, f16) loader from a row-major matrix with leading dim `ld`.
// ISA layout: lanes 0-15 row M=lane, halves[0..7]=K0..7, [8..15]=K16..23;
//             lanes 16-31 row M=lane-16, halves are K8..15 / K24..31.
__device__ __forceinline__ h16 load_a_rm(const _Float16* base, int ld,
                                         int row0, int k0, int lane) {
  const int half = lane >> 4;
  const _Float16* p = base + (size_t)(row0 + (lane & 15)) * ld + k0 + half * 8;
  h8 lo = *(const h8*)(p);        // K = k0 + 8*half + [0..7]
  h8 hi = *(const h8*)(p + 16);   // K = k0 + 8*half + 16 + [0..7]
  return cat8(lo, hi);
}

// B-matrix (32x16, f16) loader where B[k][n] = M[col0+n][k0+k] (column-of-B is
// a contiguous row of M). ISA layout: lane%16 = N, lanes 0-15 hold K=k0..7 and
// k0+16..23, lanes 16-31 hold K=k0+8..15 and k0+24..31 (matches A loader).
__device__ __forceinline__ h16 load_b_cm(const _Float16* base, int ld,
                                         int k0, int col0, int lane) {
  const _Float16* p = base + (size_t)(col0 + (lane & 15)) * ld
                    + k0 + ((lane >> 4) << 3);
  h8 lo = *(const h8*)(p);
  h8 hi = *(const h8*)(p + 16);
  return cat8(lo, hi);
}

// ---------------------------------------------------------------------------
// 1) conversion kernels
// ---------------------------------------------------------------------------
__global__ void mhca_cvt_f16(const float* __restrict__ in,
                             _Float16* __restrict__ out, int n) {
  for (int i = blockIdx.x * blockDim.x + threadIdx.x; i < n;
       i += gridDim.x * blockDim.x)
    out[i] = (_Float16)in[i];
}

// W[H][C][D] f32  ->  Wt[H][D][C] f16
__global__ void mhca_cvt_wT(const float* __restrict__ in,
                            _Float16* __restrict__ out) {
  int idx = blockIdx.x * blockDim.x + threadIdx.x;
  if (idx >= Hn * Cn * Dn) return;
  int d = idx % Dn;
  int c = (idx / Dn) % Cn;
  int h = idx / (Cn * Dn);
  out[((size_t)h * Dn + d) * Cn + c] = (_Float16)in[idx];
}

// ---------------------------------------------------------------------------
// 2) per-head projection GEMM:  Out(b,h) = In(b)[M x C] @ Wt(h)^T  (Wt=[D][C])
//    32x32 output tile per wave: grid = (M/32, D/32, B*H)
// ---------------------------------------------------------------------------
__global__ __launch_bounds__(32)
void mhca_gemm_proj(const _Float16* __restrict__ In,   // [B][M][C]
                    const _Float16* __restrict__ Wt,   // [H][D][C]
                    _Float16* __restrict__ Out,
                    int M, int transposedOut) {
  const int lane = threadIdx.x;
  const int m0 = blockIdx.x * 32;
  const int n0 = blockIdx.y * 32;
  const int bh = blockIdx.z;

  const _Float16* A  = In + (size_t)(bh / Hn) * M * Cn;
  const _Float16* Bm = Wt + (size_t)(bh % Hn) * Dn * Cn;

  f8 acc[2][2];
#pragma unroll
  for (int i = 0; i < 2; ++i)
#pragma unroll
    for (int j = 0; j < 2; ++j) { f8 z = {}; acc[i][j] = z; }

  for (int k0 = 0; k0 < Cn; k0 += 32) {
    if (k0 + 64 < Cn) {
      __builtin_prefetch((const void*)(A + (size_t)(m0 + (lane & 15)) * Cn +
                                       k0 + 64), 0, 0);
      __builtin_prefetch((const void*)(A + (size_t)(m0 + 16 + (lane & 15)) * Cn +
                                       k0 + 64), 0, 0);
    }
    const h16 a0 = load_a_rm(A, Cn, m0,      k0, lane);
    const h16 a1 = load_a_rm(A, Cn, m0 + 16, k0, lane);
    const h16 b0 = load_b_cm(Bm, Cn, k0, n0,      lane);
    const h16 b1 = load_b_cm(Bm, Cn, k0, n0 + 16, lane);
    acc[0][0] = wmma16(a0, b0, acc[0][0]);
    acc[0][1] = wmma16(a0, b1, acc[0][1]);
    acc[1][0] = wmma16(a1, b0, acc[1][0]);
    acc[1][1] = wmma16(a1, b1, acc[1][1]);
  }

  const int n  = lane & 15;
  const int rb = (lane >> 4) * 8;   // C-layout: 8 consecutive rows per lane
  if (transposedOut) {
    // V^T layout: [B*H][D][M]; 8 consecutive rows -> one contiguous b128 store
    _Float16* Ob = Out + (size_t)bh * Dn * M;
#pragma unroll
    for (int i = 0; i < 2; ++i)
#pragma unroll
      for (int j = 0; j < 2; ++j) {
        h8 v;
#pragma unroll
        for (int r = 0; r < 8; ++r) v[r] = (_Float16)acc[i][j][r];
        *(h8*)(Ob + (size_t)(n0 + j * 16 + n) * M + m0 + i * 16 + rb) = v;
      }
  } else {
    // row-major [B*H][M][D]
    _Float16* Ob = Out + (size_t)bh * M * Dn;
#pragma unroll
    for (int i = 0; i < 2; ++i)
#pragma unroll
      for (int j = 0; j < 2; ++j)
#pragma unroll
        for (int r = 0; r < 8; ++r)
          Ob[(size_t)(m0 + i * 16 + rb + r) * Dn + n0 + j * 16 + n] =
              (_Float16)acc[i][j][r];
  }
}

// ---------------------------------------------------------------------------
// 3) flash attention: one wave owns a 16-row Q tile, streams S in 32-chunks
//    grid = (T/16, B*H)
// ---------------------------------------------------------------------------
__global__ __launch_bounds__(32)
void mhca_attn(const _Float16* __restrict__ Q,   // [B*H][T][D]
               const _Float16* __restrict__ K,   // [B*H][S][D]
               const _Float16* __restrict__ Vt,  // [B*H][D][S]
               _Float16* __restrict__ O) {       // [B][T][C]
  const int lane = threadIdx.x;
  const int t0 = blockIdx.x * 16;
  const int bh = blockIdx.y;
  const int b  = bh / Hn;
  const int h  = bh % Hn;

  const _Float16* Qb = Q  + (size_t)bh * Tn * Dn;
  const _Float16* Kb = K  + (size_t)bh * Sn * Dn;
  const _Float16* Vb = Vt + (size_t)bh * Dn * Sn;
  _Float16* Ob = O + (size_t)b * Tn * Cn;

  // Q tile (16x64) held in registers as two A-fragments (K-chunks 0 and 32)
  const h16 qa0 = load_a_rm(Qb, Dn, t0, 0,  lane);
  const h16 qa1 = load_a_rm(Qb, Dn, t0, 32, lane);

  f8 oacc[4];
  float m[8], l[8];
#pragma unroll
  for (int j = 0; j < 4; ++j) { f8 z = {}; oacc[j] = z; }
#pragma unroll
  for (int r = 0; r < 8; ++r) { m[r] = -1e30f; l[r] = 0.0f; }

  __shared__ __align__(16) _Float16 pLds[16 * 32];  // P tile staging (1 KB)

  const int colb  = lane & 15;
  const int rbase = (lane >> 4) * 8;

  for (int s0 = 0; s0 < Sn; s0 += 32) {
    // ---- scores: two 16x16 tiles (cols s0..s0+15 and s0+16..s0+31) ----
    f8 z = {};
    f8 sc0 = wmma16(qa0, load_b_cm(Kb, Dn, 0,  s0,      lane), z);
    sc0    = wmma16(qa1, load_b_cm(Kb, Dn, 32, s0,      lane), sc0);
    f8 sc1 = wmma16(qa0, load_b_cm(Kb, Dn, 0,  s0 + 16, lane), z);
    sc1    = wmma16(qa1, load_b_cm(Kb, Dn, 32, s0 + 16, lane), sc1);

    float s0a[8], s1a[8], cf[8];
#pragma unroll
    for (int r = 0; r < 8; ++r) {
      s0a[r] = sc0[r] * kScale;
      s1a[r] = sc1[r] * kScale;
    }

    // ---- online softmax (row reductions across the 16-lane half-wave) ----
#pragma unroll
    for (int r = 0; r < 8; ++r) {
      float vmax = fmaxf(s0a[r], s1a[r]);
      vmax = fmaxf(vmax, __shfl_xor(vmax, 8, 16));
      vmax = fmaxf(vmax, __shfl_xor(vmax, 4, 16));
      vmax = fmaxf(vmax, __shfl_xor(vmax, 2, 16));
      vmax = fmaxf(vmax, __shfl_xor(vmax, 1, 16));
      float mn = fmaxf(m[r], vmax);
      cf[r] = __expf(m[r] - mn);
      m[r]  = mn;
      float p0 = __expf(s0a[r] - mn);
      float p1 = __expf(s1a[r] - mn);
      s0a[r] = p0;
      s1a[r] = p1;
      float rs = p0 + p1;
      rs += __shfl_xor(rs, 8, 16);
      rs += __shfl_xor(rs, 4, 16);
      rs += __shfl_xor(rs, 2, 16);
      rs += __shfl_xor(rs, 1, 16);
      l[r] = l[r] * cf[r] + rs;
    }
#pragma unroll
    for (int j = 0; j < 4; ++j)
#pragma unroll
      for (int r = 0; r < 8; ++r) oacc[j][r] *= cf[r];

    // ---- transpose P (C-layout -> A-layout) through LDS ----
    __syncthreads();
#pragma unroll
    for (int r = 0; r < 8; ++r) {
      pLds[(rbase + r) * 32 + colb]      = (_Float16)s0a[r];
      pLds[(rbase + r) * 32 + 16 + colb] = (_Float16)s1a[r];
    }
    __syncthreads();
    const h16 pa = load_a_rm(pLds, 32, 0, 0, lane);

    // ---- O += P(16x32) @ V(32x64), via V^T rows (contiguous B loads) ----
#pragma unroll
    for (int j = 0; j < 4; ++j)
      oacc[j] = wmma16(pa, load_b_cm(Vb, Sn, s0, j * 16, lane), oacc[j]);
  }

  // ---- epilogue: normalize and scatter into concat-head layout [T][H*D] ----
#pragma unroll
  for (int j = 0; j < 4; ++j)
#pragma unroll
    for (int r = 0; r < 8; ++r) {
      float o = oacc[j][r] / l[r];
      Ob[(size_t)(t0 + rbase + r) * Cn + h * Dn + j * 16 + colb] = (_Float16)o;
    }
}

// ---------------------------------------------------------------------------
// 4) output projection: out = O[BT x C] @ Wo^T + bo   (f32 result)
//    32x32 output tile per wave: grid = (BT/32, C/32)
// ---------------------------------------------------------------------------
__global__ __launch_bounds__(32)
void mhca_gemm_out(const _Float16* __restrict__ A,   // [B*T][C]
                   const _Float16* __restrict__ Wo,  // [C][C] f16 (row-major)
                   const float* __restrict__ bo,
                   float* __restrict__ out) {
  const int lane = threadIdx.x;
  const int m0 = blockIdx.x * 32;
  const int n0 = blockIdx.y * 32;

  f8 acc[2][2];
#pragma unroll
  for (int i = 0; i < 2; ++i)
#pragma unroll
    for (int j = 0; j < 2; ++j) { f8 z = {}; acc[i][j] = z; }

  for (int k0 = 0; k0 < Cn; k0 += 32) {
    if (k0 + 64 < Cn) {
      __builtin_prefetch((const void*)(A + (size_t)(m0 + (lane & 15)) * Cn +
                                       k0 + 64), 0, 0);
      __builtin_prefetch((const void*)(A + (size_t)(m0 + 16 + (lane & 15)) * Cn +
                                       k0 + 64), 0, 0);
    }
    // B[k][n] = Wo[n][k] : Wo rows give contiguous K — exactly B layout
    const h16 a0 = load_a_rm(A, Cn, m0,      k0, lane);
    const h16 a1 = load_a_rm(A, Cn, m0 + 16, k0, lane);
    const h16 b0 = load_b_cm(Wo, Cn, k0, n0,      lane);
    const h16 b1 = load_b_cm(Wo, Cn, k0, n0 + 16, lane);
    acc[0][0] = wmma16(a0, b0, acc[0][0]);
    acc[0][1] = wmma16(a0, b1, acc[0][1]);
    acc[1][0] = wmma16(a1, b0, acc[1][0]);
    acc[1][1] = wmma16(a1, b1, acc[1][1]);
  }

  const int n  = lane & 15;
  const int rb = (lane >> 4) * 8;
#pragma unroll
  for (int j = 0; j < 2; ++j) {
    const float bias = bo[n0 + j * 16 + n];
#pragma unroll
    for (int i = 0; i < 2; ++i)
#pragma unroll
      for (int r = 0; r < 8; ++r)
        out[(size_t)(m0 + i * 16 + rb + r) * Cn + n0 + j * 16 + n] =
            acc[i][j][r] + bias;
  }
}

// ---------------------------------------------------------------------------
extern "C" void kernel_launch(void* const* d_in, const int* in_sizes, int n_in,
                              void* d_out, int out_size, void* d_ws,
                              size_t ws_size, hipStream_t stream) {
  (void)in_sizes; (void)n_in; (void)out_size; (void)ws_size;

  const float* x    = (const float*)d_in[0];  // [B,T,C]
  const float* y    = (const float*)d_in[1];  // [B,S,C]
  const float* Wq   = (const float*)d_in[2];  // [H,C,D]
  const float* Wk   = (const float*)d_in[3];
  const float* Wv   = (const float*)d_in[4];
  const float* Wo   = (const float*)d_in[5];  // [C,C]
  const float* bo   = (const float*)d_in[6];  // [C]
  float* out        = (float*)d_out;          // [B,T,C]

  // workspace carve-up (all f16)
  char* w = (char*)d_ws;
  auto take = [&](size_t elems) {
    _Float16* p = (_Float16*)w;
    w += elems * sizeof(_Float16);
    return p;
  };
  _Float16* Xh   = take((size_t)Bn * Tn * Cn);   // 8 MB
  _Float16* Yh   = take((size_t)Bn * Sn * Cn);   // 8 MB
  _Float16* Wqt  = take((size_t)Hn * Dn * Cn);   // 2 MB
  _Float16* Wkt  = take((size_t)Hn * Dn * Cn);   // 2 MB
  _Float16* Wvt  = take((size_t)Hn * Dn * Cn);   // 2 MB
  _Float16* Woh  = take((size_t)Cn * Cn);        // 2 MB
  _Float16* Qh   = take((size_t)Bn * Hn * Tn * Dn);  // 8 MB
  _Float16* Kh   = take((size_t)Bn * Hn * Sn * Dn);  // 8 MB
  _Float16* Vth  = take((size_t)Bn * Hn * Dn * Sn);  // 8 MB
  _Float16* Oh   = take((size_t)Bn * Tn * Cn);       // 8 MB

  // ---- stage 1: precision conversion / weight transposes ----
  {
    const int nBTC = Bn * Tn * Cn;
    mhca_cvt_f16<<<(nBTC + 255) / 256, 256, 0, stream>>>(x, Xh, nBTC);
    mhca_cvt_f16<<<(nBTC + 255) / 256, 256, 0, stream>>>(y, Yh, nBTC);
    const int nCC = Cn * Cn;
    mhca_cvt_f16<<<(nCC + 255) / 256, 256, 0, stream>>>(Wo, Woh, nCC);
    const int nW = Hn * Cn * Dn;
    mhca_cvt_wT<<<(nW + 255) / 256, 256, 0, stream>>>(Wq, Wqt);
    mhca_cvt_wT<<<(nW + 255) / 256, 256, 0, stream>>>(Wk, Wkt);
    mhca_cvt_wT<<<(nW + 255) / 256, 256, 0, stream>>>(Wv, Wvt);
  }

  // ---- stage 2: Q / K / V projections (V stored transposed) ----
  {
    dim3 grid(Tn / 32, Dn / 32, Bn * Hn);
    mhca_gemm_proj<<<grid, 32, 0, stream>>>(Xh, Wqt, Qh,  Tn, 0);
    mhca_gemm_proj<<<grid, 32, 0, stream>>>(Yh, Wkt, Kh,  Sn, 0);
    mhca_gemm_proj<<<grid, 32, 0, stream>>>(Yh, Wvt, Vth, Sn, 1);
  }

  // ---- stage 3: flash attention ----
  {
    dim3 grid(Tn / 16, Bn * Hn);
    mhca_attn<<<grid, 32, 0, stream>>>(Qh, Kh, Vth, Oh);
  }

  // ---- stage 4: output projection + bias ----
  {
    dim3 grid((Bn * Tn) / 32, Cn / 32);
    mhca_gemm_out<<<grid, 32, 0, stream>>>(Oh, Woh, bo, out);
  }
}
